// MultiHeadedAttention_25305947308727
// MI455X (gfx1250) — matrix operand order repaired and checked by reading, not compiled, not run on hardware
//
#include <hip/hip_runtime.h>
#include <hip/hip_bf16.h>

// ---------------------------------------------------------------------------
// MI455X (gfx1250) GQA attention layer, bf16 WMMA + Tensor Data Mover.
//   B=2, S=1024, HID=2048, H=32, D=64, KV=8 (GQA rep=4)
// Pipeline: weight transpose->bf16, QKV projection GEMMs (WMMA, TDM-staged B),
//           fused flash attention (TDM-staged K, WMMA QK^T + online softmax +
//           WMMA PV), output projection GEMM (WMMA, f32 out + bias).
// ---------------------------------------------------------------------------

typedef __bf16 bh16;
typedef __attribute__((ext_vector_type(16))) __bf16 v16bf;
typedef __attribute__((ext_vector_type(8)))  float  v8f;

#define S_LEN  1024
#define BATCH  2
#define NHEAD  32
#define HDIM   64
#define KVH    8
#define HIDN   2048
#define KVDIM  512

union FragU { v16bf v; uint4 u[2]; };
union Pack8 { bh16 h[8]; uint4 u; };

__device__ __forceinline__ v8f wmma_bf16(v16bf a, v16bf b, v8f c) {
  // D = A(16x32 bf16) * B(32x16 bf16) + C(16x16 f32)
  return __builtin_amdgcn_wmma_f32_16x16x32_bf16(
      false, a, false, b, (short)0, c, false, false);
}

// A-fragment (16x32 bf16) from row-major [m][k] source, leading dim ld (elems).
__device__ __forceinline__ v16bf load_frag_a(const bh16* p, int ld) {
  const int lane = threadIdx.x & 31;
  const int m = lane & 15, hf = lane >> 4;
  const bh16* r = p + (size_t)m * ld + hf * 8;
  FragU f;
  f.u[0] = *(const uint4*)(r);
  f.u[1] = *(const uint4*)(r + 16);
  return f.v;
}

// B-fragment (32x16 bf16) from B^T stored row-major [n][k], leading dim ld.
__device__ __forceinline__ v16bf load_frag_b(const bh16* p, int ld) {
  const int lane = threadIdx.x & 31;
  const int n = lane & 15, hf = lane >> 4;
  const bh16* r = p + (size_t)n * ld + hf * 8;
  FragU f;
  f.u[0] = *(const uint4*)(r);
  f.u[1] = *(const uint4*)(r + 16);
  return f.v;
}

// ---------------------------------------------------------------------------
// Tensor Data Mover: async 2D bf16 tile load global -> LDS with row padding.
// Descriptor per CDNA5 ISA ch.8 (D# groups). Issued once per wave; completion
// via TENSORcnt. Guarded: builtin arity differs clang-22 (5) vs clang-23 (6).
// ---------------------------------------------------------------------------
#if __has_builtin(__builtin_amdgcn_tensor_load_to_lds)
#define HAVE_TDM 1
#else
#define HAVE_TDM 0
#endif

#if HAVE_TDM
typedef __attribute__((ext_vector_type(4))) unsigned int u32x4;
typedef __attribute__((ext_vector_type(4))) int i32x4;
typedef __attribute__((ext_vector_type(8))) int i32x8;

// tile_w elems/row (bf16), tile_h rows, row stride in elems.
// pad_iv: pad_interval code (DWORDs before pad = 2<<code)
// pad_am: pad_amount code (pad DWORDs = code+1)
__device__ __forceinline__ void tdm_load_2d(unsigned int lds_off,
                                            const void* gaddr,
                                            unsigned int tile_w,
                                            unsigned int tile_h,
                                            unsigned int stride_elems,
                                            unsigned int pad_iv,
                                            unsigned int pad_am) {
  const unsigned long long ga = (unsigned long long)(size_t)gaddr;
  u32x4 g0;
  g0[0] = 1u;                                      // count=1 (valid user D#)
  g0[1] = lds_off;                                 // LDS byte address
  g0[2] = (unsigned int)(ga & 0xFFFFFFFFull);      // global_addr[31:0]
  g0[3] = (unsigned int)((ga >> 32) & 0x1FFFFFFull) | (2u << 30); // type=2
  i32x8 g1;
  g1[0] = (int)((1u << 16)            // data_size = 1 -> 2 bytes
              | (1u << 20)            // pad_enable
              | (pad_iv << 22)        // pad_interval
              | (pad_am << 25));      // pad_amount
  g1[1] = (int)(tile_w << 16);        // tensor_dim0[15:0] (== tile_w, no OOB)
  g1[2] = (int)(tile_h << 16);        // tensor_dim0 hi=0 | tensor_dim1[15:0]
  g1[3] = (int)(tile_w << 16);        // tensor_dim1 hi=0 | tile_dim0
  g1[4] = (int)tile_h;                // tile_dim1 | tile_dim2=0 (2D)
  g1[5] = (int)stride_elems;          // tensor_dim0_stride[31:0] (elems)
  g1[6] = 0;                          // stride hi | dim1_stride lo (unused 2D)
  g1[7] = 0;
  const i32x4 z4 = {0, 0, 0, 0};      // groups 2/3: tile_dim3/4 = 0 (unused)
#if defined(__clang_major__) && (__clang_major__ >= 23)
  const i32x8 z8 = {0, 0, 0, 0, 0, 0, 0, 0};
  __builtin_amdgcn_tensor_load_to_lds(g0, g1, z4, z4, z8, 0);
#else
  __builtin_amdgcn_tensor_load_to_lds(g0, g1, z4, z4, 0);
#endif
}
#endif // HAVE_TDM

// ---------------------------------------------------------------------------
// Weight pre-pass: W f32 [K][N]  ->  Wt bf16 [N][K]  (coalesced both sides)
// ---------------------------------------------------------------------------
__global__ __launch_bounds__(256)
void transpose_cvt(const float* __restrict__ W, bh16* __restrict__ Wt,
                   int K, int N) {
  __shared__ float tile[32][33];
  const int tx = threadIdx.x, ty = threadIdx.y;   // block (32,8)
  const int bx = blockIdx.x, by = blockIdx.y;     // bx over N, by over K
#pragma unroll
  for (int j = 0; j < 4; ++j) {
    const int k = by * 32 + ty + j * 8;
    const int n = bx * 32 + tx;
    tile[ty + j * 8][tx] = W[(size_t)k * N + n];
  }
  __syncthreads();
#pragma unroll
  for (int j = 0; j < 4; ++j) {
    const int n = bx * 32 + ty + j * 8;
    const int k = by * 32 + tx;
    Wt[(size_t)n * K + k] = (bh16)tile[tx][ty + j * 8];
  }
}

// ---------------------------------------------------------------------------
// WMMA GEMM: C[M,N] = A[M,K] * Bt[N,K]^T + bias.
// Block = 256 thr (8 waves), tile 128x128, BK=32.
// B tile staged by TDM (async DMA w/ hardware row padding) while the VALU
// lanes stage/convert the A tile; wave 0 then waits TENSORcnt and barriers.
// ---------------------------------------------------------------------------
template<bool A_IS_F32, bool OUT_IS_F32>
__global__ __launch_bounds__(256)
void gemm_wmma(const void* __restrict__ Ap, const bh16* __restrict__ Bt,
               const float* __restrict__ bias, void* __restrict__ Cp,
               int M, int N, int K) {
  __shared__ bh16 As[128 * 40];   // rows padded 32->40 (bank spread)
  __shared__ bh16 Bs[128 * 40];
  const int bn = blockIdx.x * 128;
  const int bm = blockIdx.y * 128;
  const int t = threadIdx.x;
  const int wave = t >> 5, lane = t & 31;
  const int wm = wave >> 2, wn = wave & 3;
  const int srow = t >> 1;            // 0..127
  const int scb = (t & 1) * 16;       // 0 or 16

  v8f acc[4][2] = {};

  for (int k0 = 0; k0 < K; k0 += 32) {
    __syncthreads();   // previous tile consumed; safe to overwrite LDS

#if HAVE_TDM
    // ---- B tile (128 n-rows x 32 k) via Tensor Data Mover ----
    // rows: 32 bf16 = 16 DW (pad_interval code 3), pad 4 DW (code 3) -> ld 40
    if (wave == 0)
      tdm_load_2d((unsigned int)(size_t)&Bs[0],
                  Bt + (size_t)bn * K + k0,
                  32u, 128u, (unsigned int)K, 3u, 3u);
#else
    {
      const bh16* bp = Bt + (size_t)(bn + srow) * K + k0 + scb;
      *(uint4*)&Bs[srow * 40 + scb]     = ((const uint4*)bp)[0];
      *(uint4*)&Bs[srow * 40 + scb + 8] = ((const uint4*)bp)[1];
    }
#endif

    // ---- stage A tile (128 x 32), convert f32->bf16 if needed ----
    if (A_IS_F32) {
      const float* ap = (const float*)Ap + (size_t)(bm + srow) * K + k0 + scb;
      float xs[16];
      ((float4*)xs)[0] = ((const float4*)ap)[0];
      ((float4*)xs)[1] = ((const float4*)ap)[1];
      ((float4*)xs)[2] = ((const float4*)ap)[2];
      ((float4*)xs)[3] = ((const float4*)ap)[3];
      Pack8 pk0, pk1;
#pragma unroll
      for (int j = 0; j < 8; ++j) pk0.h[j] = (bh16)xs[j];
#pragma unroll
      for (int j = 0; j < 8; ++j) pk1.h[j] = (bh16)xs[8 + j];
      *(uint4*)&As[srow * 40 + scb]     = pk0.u;
      *(uint4*)&As[srow * 40 + scb + 8] = pk1.u;
      if (k0 + 32 < K) __builtin_prefetch(ap + 32, 0, 1);
    } else {
      const bh16* ap = (const bh16*)Ap + (size_t)(bm + srow) * K + k0 + scb;
      *(uint4*)&As[srow * 40 + scb]     = ((const uint4*)ap)[0];
      *(uint4*)&As[srow * 40 + scb + 8] = ((const uint4*)ap)[1];
      if (k0 + 32 < K) __builtin_prefetch(ap + 32, 0, 1);
    }

#if HAVE_TDM
    if (wave == 0) __builtin_amdgcn_s_wait_tensorcnt(0);
#endif
    __syncthreads();   // tiles published

    const v16bf bf0 = load_frag_b(&Bs[(wn * 32 +  0) * 40], 40);
    const v16bf bf1 = load_frag_b(&Bs[(wn * 32 + 16) * 40], 40);
#pragma unroll
    for (int mt = 0; mt < 4; ++mt) {
      const v16bf af = load_frag_a(&As[(wm * 64 + mt * 16) * 40], 40);
      acc[mt][0] = wmma_bf16(af, bf0, acc[mt][0]);
      acc[mt][1] = wmma_bf16(af, bf1, acc[mt][1]);
    }
  }

  // ---- epilogue: C layout: VGPR r -> row r (+8 for upper lane half) ----
  const int ln = lane & 15, hf = lane >> 4;
#pragma unroll
  for (int mt = 0; mt < 4; ++mt)
#pragma unroll
    for (int nt = 0; nt < 2; ++nt)
#pragma unroll
      for (int r = 0; r < 8; ++r) {
        const int row = bm + wm * 64 + mt * 16 + hf * 8 + r;
        const int col = bn + wn * 32 + nt * 16 + ln;
        const float v = acc[mt][nt][r] + bias[col];
        if (OUT_IS_F32) ((float*)Cp)[(size_t)row * N + col] = v;
        else            ((bh16*)Cp)[(size_t)row * N + col] = (bh16)v;
      }
}

// ---------------------------------------------------------------------------
// Fused flash attention (bf16 WMMA, f32 accumulate, online softmax).
// grid = (S/64, H, B), block = 128 thr (4 waves). Wave w owns 16 query rows.
// K tile staged by TDM; V tile transposed manually by all lanes in parallel.
// ---------------------------------------------------------------------------
__global__ __launch_bounds__(128)
void attn_flash(const bh16* __restrict__ Q, const bh16* __restrict__ Kp,
                const bh16* __restrict__ Vp, const float* __restrict__ pb,
                const float* __restrict__ mask, bh16* __restrict__ Ctx) {
  __shared__ bh16 Ks[64 * 72];      // [key][d]   (B^T layout for QK^T)
  __shared__ bh16 Vt[64 * 72];      // [d][key]   (B^T layout for P*V)
  __shared__ bh16 Ps[4][16 * 72];   // per-wave P scratch (C-layout -> A-layout)

  const int qt = blockIdx.x, h = blockIdx.y, b = blockIdx.z;
  const int t = threadIdx.x;
  const int wave = t >> 5, lane = t & 31;
  const int ln = lane & 15, hf = lane >> 4;
  const int kv = h >> 2;                      // GQA: rep = H/KV = 4
  const int q0 = qt * 64 + wave * 16;

  // Q fragments for this wave's 16 rows (held for the whole kernel)
  const bh16* qrow = Q + (size_t)(b * S_LEN + q0) * HIDN + h * HDIM;
  const v16bf qa0 = load_frag_a(qrow, HIDN);        // d = 0..31
  const v16bf qa1 = load_frag_a(qrow + 32, HIDN);   // d = 32..63

  float rmax[8], rsum[8];
#pragma unroll
  for (int r = 0; r < 8; ++r) { rmax[r] = -3.0e38f; rsum[r] = 0.f; }
  v8f O[4] = {};

  const int rr = t >> 1;          // 0..63 : staged row
  const int db = (t & 1) * 32;    // 0/32  : staged half of D

  for (int kt = 0; kt < S_LEN / 64; ++kt) {
    __syncthreads();   // previous K/V tiles consumed
    const size_t src = (size_t)(b * S_LEN + kt * 64 + rr) * KVDIM
                     + (size_t)kv * HDIM + db;
#if HAVE_TDM
    // ---- K tile (64 keys x 64 d) via TDM ----
    // rows: 64 bf16 = 32 DW (pad_interval code 4), pad 4 DW (code 3) -> ld 72
    if (wave == 0)
      tdm_load_2d((unsigned int)(size_t)&Ks[0],
                  Kp + (size_t)(b * S_LEN + kt * 64) * KVDIM + kv * HDIM,
                  64u, 64u, (unsigned int)KVDIM, 4u, 3u);
#else
    {
      const uint4* kp = (const uint4*)(Kp + src);
      uint4* kd = (uint4*)&Ks[rr * 72 + db];
      kd[0] = kp[0]; kd[1] = kp[1]; kd[2] = kp[2]; kd[3] = kp[3];
    }
#endif
    // ---- V staged transposed: Vt[d][key] (all 128 lanes) ----
    {
      union { uint4 u[4]; bh16 h[32]; } vv;
      const uint4* vp = (const uint4*)(Vp + src);
      vv.u[0] = vp[0]; vv.u[1] = vp[1]; vv.u[2] = vp[2]; vv.u[3] = vp[3];
#pragma unroll
      for (int j = 0; j < 32; ++j) Vt[(size_t)(db + j) * 72 + rr] = vv.h[j];
    }
#if HAVE_TDM
    if (wave == 0) __builtin_amdgcn_s_wait_tensorcnt(0);
#endif
    __syncthreads();   // tiles published

    // ---- S tile = Q (16x64) * K^T (64x64): 4 n-tiles x 2 K-steps ----
    v8f sc[4];
#pragma unroll
    for (int nt = 0; nt < 4; ++nt) {
      v8f z = {};
      z = wmma_bf16(qa0, load_frag_b(&Ks[nt * 16 * 72], 72), z);
      z = wmma_bf16(qa1, load_frag_b(&Ks[nt * 16 * 72 + 32], 72), z);
      sc[nt] = z;
    }

    // ---- bias + scale + mask in f32 accumulators ----
#pragma unroll
    for (int nt = 0; nt < 4; ++nt)
#pragma unroll
      for (int r = 0; r < 8; ++r) {
        const int qg = q0 + hf * 8 + r;
        const int cg = kt * 64 + nt * 16 + ln;
        sc[nt][r] = (sc[nt][r] + pb[((size_t)h * S_LEN + qg) * S_LEN + cg]) * 0.125f
                    + mask[((size_t)b * S_LEN + qg) * S_LEN + cg];
      }

    // ---- online softmax (rows live in 16-lane halves of the wave) ----
    bh16* ps = &Ps[wave][0];
#pragma unroll
    for (int r = 0; r < 8; ++r) {
      float m = fmaxf(fmaxf(sc[0][r], sc[1][r]), fmaxf(sc[2][r], sc[3][r]));
#pragma unroll
      for (int off = 8; off >= 1; off >>= 1)
        m = fmaxf(m, __shfl_xor(m, off, 32));
      const float nm = fmaxf(rmax[r], m);
      float srow = 0.f;
#pragma unroll
      for (int nt = 0; nt < 4; ++nt) {
        const float p = __expf(sc[nt][r] - nm);
        sc[nt][r] = p;
        srow += p;
      }
#pragma unroll
      for (int off = 8; off >= 1; off >>= 1)
        srow += __shfl_xor(srow, off, 32);
      const float fac = __expf(rmax[r] - nm);
      rsum[r] = rsum[r] * fac + srow;
      rmax[r] = nm;
#pragma unroll
      for (int dt = 0; dt < 4; ++dt) O[dt][r] *= fac;   // rescale before accum
      // spill P (C-layout) into LDS so it can be reloaded as A fragments
#pragma unroll
      for (int nt = 0; nt < 4; ++nt)
        ps[(hf * 8 + r) * 72 + nt * 16 + ln] = (bh16)sc[nt][r];
    }

    // ---- O += P (16x64) * V (64x64) ----  (DS ops are in-order per wave)
    const v16bf pa0 = load_frag_a(ps, 72);
    const v16bf pa1 = load_frag_a(ps + 32, 72);
#pragma unroll
    for (int dt = 0; dt < 4; ++dt) {
      O[dt] = wmma_bf16(pa0, load_frag_b(&Vt[dt * 16 * 72], 72), O[dt]);
      O[dt] = wmma_bf16(pa1, load_frag_b(&Vt[dt * 16 * 72 + 32], 72), O[dt]);
    }
  }

  // ---- normalize and write context bf16 [B*S][HID] (col = h*64 + d) ----
#pragma unroll
  for (int dt = 0; dt < 4; ++dt)
#pragma unroll
    for (int r = 0; r < 8; ++r) {
      const int qg = q0 + hf * 8 + r;
      Ctx[(size_t)(b * S_LEN + qg) * HIDN + h * HDIM + dt * 16 + ln] =
          (bh16)(O[dt][r] / rsum[r]);
    }
}

// ---------------------------------------------------------------------------
// Host-side launch. Workspace (~40 MB): WqT, WkT, WvT, WoT, Qbf, Kbf, Vbf, Ctx
// ---------------------------------------------------------------------------
extern "C" void kernel_launch(void* const* d_in, const int* in_sizes, int n_in,
                              void* d_out, int out_size, void* d_ws, size_t ws_size,
                              hipStream_t stream) {
  (void)in_sizes; (void)n_in; (void)out_size; (void)ws_size;
  const float* query = (const float*)d_in[0];
  const float* key   = (const float*)d_in[1];
  const float* value = (const float*)d_in[2];
  const float* mask  = (const float*)d_in[3];
  const float* pb    = (const float*)d_in[4];
  const float* Wq    = (const float*)d_in[5];
  const float* bq    = (const float*)d_in[6];
  const float* Wk    = (const float*)d_in[7];
  const float* bk    = (const float*)d_in[8];
  const float* Wv    = (const float*)d_in[9];
  const float* bv    = (const float*)d_in[10];
  const float* Wo    = (const float*)d_in[11];
  const float* bo    = (const float*)d_in[12];
  float* out = (float*)d_out;

  char* w = (char*)d_ws;
  bh16* WqT = (bh16*)w; w += (size_t)HIDN  * HIDN * 2;
  bh16* WkT = (bh16*)w; w += (size_t)KVDIM * HIDN * 2;
  bh16* WvT = (bh16*)w; w += (size_t)KVDIM * HIDN * 2;
  bh16* WoT = (bh16*)w; w += (size_t)HIDN  * HIDN * 2;
  bh16* Qb  = (bh16*)w; w += (size_t)BATCH * S_LEN * HIDN  * 2;
  bh16* Kb  = (bh16*)w; w += (size_t)BATCH * S_LEN * KVDIM * 2;
  bh16* Vb  = (bh16*)w; w += (size_t)BATCH * S_LEN * KVDIM * 2;
  bh16* Ctx = (bh16*)w; w += (size_t)BATCH * S_LEN * HIDN  * 2;

  dim3 tb(32, 8);
  transpose_cvt<<<dim3(HIDN / 32,  HIDN / 32), tb, 0, stream>>>(Wq, WqT, HIDN, HIDN);
  transpose_cvt<<<dim3(KVDIM / 32, HIDN / 32), tb, 0, stream>>>(Wk, WkT, HIDN, KVDIM);
  transpose_cvt<<<dim3(KVDIM / 32, HIDN / 32), tb, 0, stream>>>(Wv, WvT, HIDN, KVDIM);
  transpose_cvt<<<dim3(HIDN / 32,  HIDN / 32), tb, 0, stream>>>(Wo, WoT, HIDN, HIDN);

  const int M = BATCH * S_LEN;  // 2048 rows
  gemm_wmma<true,  false><<<dim3(HIDN / 128,  M / 128), 256, 0, stream>>>(
      query, WqT, bq, Qb, M, HIDN, HIDN);
  gemm_wmma<true,  false><<<dim3(KVDIM / 128, M / 128), 256, 0, stream>>>(
      key, WkT, bk, Kb, M, KVDIM, HIDN);
  gemm_wmma<true,  false><<<dim3(KVDIM / 128, M / 128), 256, 0, stream>>>(
      value, WvT, bv, Vb, M, KVDIM, HIDN);

  attn_flash<<<dim3(S_LEN / 64, NHEAD, BATCH), 128, 0, stream>>>(
      Qb, Kb, Vb, pb, mask, Ctx);

  gemm_wmma<false, true><<<dim3(HIDN / 128, M / 128), 256, 0, stream>>>(
      Ctx, WoT, bo, out, M, HIDN, HIDN);
}
